// MyLSTM_44032004719093
// MI455X (gfx1250) — matrix operand order repaired and checked by reading, not compiled, not run on hardware
//
#include <hip/hip_runtime.h>

// ---------------------------------------------------------------------------
// Reference: 2-layer "scalar-gate" LSTM. S=1024, B=128, H=1024, D=2H=2048.
// Gates are [B,4] scalars broadcast over [B,H].
//
//   Kernel 1 (gx_kernel): Gx[t,b,j] = sum_k x[t,b,k] * W[0][j][H+k]
//     - fully parallel, bandwidth-bound (reads 512MB X once, writes 2MB).
//   Kernel 2 (lstm_rec_kernel): the sequential recurrence.
//     - 16 blocks x 512 threads; each block owns 8 batch rows for all 1024
//       steps. State (c f32, h f16) lives in LDS (~125KB dynamic).
//     - Gate matmuls via v_wmma_f32_16x16x32_f16 (M=16 batch tile, 4 live
//       gate columns of N=16, K split across 16 waves, LDS reduction).
//     - Out-of-range WMMA rows/cols read a 64B zeroed LDS buffer through a
//       cndmask'd address: keeps execution convergent (no exec-mask dance).
//     - Gx prefetched one step ahead (regs) + 8 ahead (global_prefetch_b8).
// ---------------------------------------------------------------------------

#define S_STEPS 1024
#define B_N     128
#define H_N     1024
#define ROWS    8      // batch rows per recurrent block
#define THREADS 512    // 16 wave32s
#define NWAVES  16

typedef __attribute__((ext_vector_type(16))) _Float16 v16h;
typedef __attribute__((ext_vector_type(8)))  _Float16 v8h;
typedef __attribute__((ext_vector_type(8)))  float    v8f;

// ---------------------------------------------------------------------------
// One 16x16x32 f16 WMMA accumulation.
//   A (16x32 f16): row M = lane&15 from LDS h-state (f16), K window [kh,kh+32)
//     ISA layout: lanes<16 hold K offs {0..7,16..23}; lanes>=16 {8..15,24..31}.
//   B (32x16 f16): col N = lane&15 (only N<4 = gates live, rest zero),
//     same K layout as A (lanes<16 {0..7,16..23}; lanes>=16 {8..15,24..31}), from Wt[gate][K].
//   Invalid rows/cols read from zbuf (32 zeroed halves) - address select only.
// ---------------------------------------------------------------------------
__device__ __forceinline__ v8f wmma_step(v8f acc,
    const _Float16* __restrict__ hsrc, int kh,
    const _Float16* __restrict__ Wt, int wstride, int kw,
    int lane, const _Float16* __restrict__ zbuf)
{
  const int rn    = lane & 15;               // A row / B column
  const int hi16  = (lane < 16) ? 0 : 1;

  const _Float16* hp = (rn < ROWS) ? (hsrc + rn * H_N + kh + hi16 * 8) : zbuf;
  const v8h alo = *(const v8h*)(hp);
  const v8h ahi = *(const v8h*)(hp + 16);

  const _Float16* wp = (rn < 4) ? (Wt + rn * wstride + kw + hi16 * 8) : zbuf;
  const v8h blo = *(const v8h*)(wp);
  const v8h bhi = *(const v8h*)(wp + 16);

  v16h a, b;
#pragma unroll
  for (int e = 0; e < 8; ++e) {
    a[e] = alo[e]; a[e + 8] = ahi[e];
    b[e] = blo[e]; b[e + 8] = bhi[e];
  }
  // (neg_a, A, neg_b, B, c_mod, C, reuse_a, reuse_b)
  return __builtin_amdgcn_wmma_f32_16x16x32_f16(false, a, false, b,
                                                (short)0, acc, false, false);
}

// ---------------------------------------------------------------------------
// Kernel 1: Gx[t,b,:] = x[t,b,:] @ Wx0^T   (Wx0[j][k] = W[0][j][1024+k])
// 256 threads = 8 waves; one (t,b) row per wave; coalesced float4 loads.
// ---------------------------------------------------------------------------
__global__ void __launch_bounds__(256)
gx_kernel(const float* __restrict__ X, const float* __restrict__ Wg,
          float* __restrict__ Gx)
{
  __shared__ float Wx[4][H_N];             // 16 KB
  const int tid = threadIdx.x;
  for (int e = tid; e < 4 * H_N; e += 256) {
    int j = e >> 10, k = e & (H_N - 1);
    Wx[j][k] = Wg[j * 2048 + H_N + k];     // x-half of layer-0 rows
  }
  __syncthreads();

  const int wave = tid >> 5, lane = tid & 31;
  const long long row = (long long)blockIdx.x * 8 + wave;   // t*B + b
  const float4* xr = (const float4*)(X + row * (long long)H_N);

  float a0 = 0.f, a1 = 0.f, a2 = 0.f, a3 = 0.f;
#pragma unroll
  for (int it = 0; it < 8; ++it) {
    const int idx = lane + 32 * it;        // float4 index
    const float4 xv = xr[idx];
    const int k = idx * 4;
    const float4 w0 = *(const float4*)&Wx[0][k];
    const float4 w1 = *(const float4*)&Wx[1][k];
    const float4 w2 = *(const float4*)&Wx[2][k];
    const float4 w3 = *(const float4*)&Wx[3][k];
    a0 += xv.x * w0.x + xv.y * w0.y + xv.z * w0.z + xv.w * w0.w;
    a1 += xv.x * w1.x + xv.y * w1.y + xv.z * w1.z + xv.w * w1.w;
    a2 += xv.x * w2.x + xv.y * w2.y + xv.z * w2.z + xv.w * w2.w;
    a3 += xv.x * w3.x + xv.y * w3.y + xv.z * w3.z + xv.w * w3.w;
  }
#pragma unroll
  for (int m = 16; m >= 1; m >>= 1) {
    a0 += __shfl_xor(a0, m, 32);
    a1 += __shfl_xor(a1, m, 32);
    a2 += __shfl_xor(a2, m, 32);
    a3 += __shfl_xor(a3, m, 32);
  }
  if (lane == 0) ((float4*)Gx)[row] = make_float4(a0, a1, a2, a3);
}

// ---------------------------------------------------------------------------
// Kernel 2: the recurrence. One block = 8 batch rows for all 1024 steps.
// ---------------------------------------------------------------------------
__global__ void __launch_bounds__(THREADS)
lstm_rec_kernel(const float* __restrict__ Wg, const float* __restrict__ bg,
                const float* __restrict__ Gx, float* __restrict__ out)
{
  extern __shared__ char smem[];
  float*    c0      = (float*)(smem);              // [8][1024] f32
  float*    c1      = c0 + ROWS * H_N;             // [8][1024] f32
  float*    partial = c1 + ROWS * H_N;             // [16 waves][4 lanes][8 f32]
  float*    Gates   = partial + NWAVES * 32;       // [8][4] post-activation
  float*    gxbuf   = Gates + 32;                  // [8][4]
  float*    biasL   = gxbuf + 32;                  // [2][4]
  _Float16* zbuf    = (_Float16*)(biasL + 8);      // 32 zeroed halves (64 B)
  _Float16* h0f     = zbuf + 32;                   // [8][1024] f16 (A-matrix src)
  _Float16* h1f     = h0f + ROWS * H_N;            // [8][1024] f16
  _Float16* W0t     = h1f + ROWS * H_N;            // [4][1024] f16 (layer0 h-part)
  _Float16* W1t     = W0t + 4 * H_N;               // [4][2048] f16 (layer1 full)

  const int tid   = threadIdx.x;
  const int wave  = tid >> 5;
  const int lane  = tid & 31;
  const int bTile = blockIdx.x * ROWS;

  // ---- init state + stage weights (f16) ----
  for (int e = tid; e < ROWS * H_N; e += THREADS) {
    c0[e] = 0.f; c1[e] = 0.f;
    h0f[e] = (_Float16)0.0f; h1f[e] = (_Float16)0.0f;
  }
  for (int e = tid; e < 4 * H_N; e += THREADS) {           // layer0 h-part
    int j = e >> 10, k = e & (H_N - 1);
    W0t[e] = (_Float16)Wg[j * 2048 + k];
  }
  for (int e = tid; e < 4 * 2048; e += THREADS) {          // layer1 full [h1|h0]
    int j = e >> 11, k = e & 2047;
    W1t[e] = (_Float16)Wg[8192 + j * 2048 + k];
  }
  if (tid < 8) biasL[tid] = bg[tid];
  if (tid < 32) zbuf[tid] = (_Float16)0.0f;

  float4 gxr = make_float4(0.f, 0.f, 0.f, 0.f);
  if (tid < ROWS) gxr = ((const float4*)Gx)[bTile + tid];  // t = 0
  __syncthreads();

  const size_t OUT_HT = (size_t)S_STEPS * B_N * H_N;
  const size_t OUT_CT = OUT_HT + (size_t)2 * B_N * H_N;

  for (int t = 0; t < S_STEPS; ++t) {
    if (tid < ROWS) ((float4*)gxbuf)[tid] = gxr;
    __syncthreads();                                       // B0: gxbuf ready

    // prefetch Gx for later steps (latency fully hidden by the step body)
    if (tid < ROWS && t + 1 < S_STEPS)
      gxr = ((const float4*)Gx)[(size_t)(t + 1) * B_N + bTile + tid];
    if (tid < ROWS && t + 8 < S_STEPS)
      __builtin_prefetch(&((const float4*)Gx)[(size_t)(t + 8) * B_N + bTile + tid], 0, 1);

    // ---- layer 0 gate matmul: K=1024 over h0, 64 K per wave ----
    {
      v8f acc = {};
      const int kb = wave * 64;
      acc = wmma_step(acc, h0f, kb,      W0t, H_N, kb,      lane, zbuf);
      acc = wmma_step(acc, h0f, kb + 32, W0t, H_N, kb + 32, lane, zbuf);
      if (lane < 4) {                    // lanes 0..3 hold rows 0..7 of col N=lane
        float* pp = partial + wave * 32 + lane * 8;
        *(float4*)(pp)     = make_float4(acc[0], acc[1], acc[2], acc[3]);
        *(float4*)(pp + 4) = make_float4(acc[4], acc[5], acc[6], acc[7]);
      }
    }
    __syncthreads();                                       // B1: partials ready
    if (tid < 32) {                                        // r=tid>>2, j=tid&3
      const int r = tid >> 2, j = tid & 3;
      float g = biasL[j] + gxbuf[tid];
#pragma unroll
      for (int w = 0; w < NWAVES; ++w) g += partial[w * 32 + j * 8 + r];
      Gates[tid] = (j == 1) ? tanhf(g) : 1.f / (1.f + __expf(-g));
    }
    __syncthreads();                                       // B2: gates ready
    // ---- layer 0 elementwise: c0, h0 ----
#pragma unroll 4
    for (int i = 0; i < ROWS * H_N / THREADS; ++i) {
      const int e = tid + THREADS * i;
      const int r = e >> 10;
      const float ii = Gates[r * 4 + 0], ch = Gates[r * 4 + 1];
      const float ff = Gates[r * 4 + 2], oo = Gates[r * 4 + 3];
      const float cn = ff * c0[e] + ii * ch;
      c0[e] = cn;
      const float hn = oo * tanhf(cn);
      h0f[e] = (_Float16)hn;
      if (t == S_STEPS - 1) {
        out[OUT_HT + (size_t)bTile * H_N + e] = hn;        // hT layer 0
        out[OUT_CT + (size_t)bTile * H_N + e] = cn;        // cT layer 0
      }
    }
    __syncthreads();                                       // B3: h0f ready

    // ---- layer 1 gate matmul: K=2048 over z=[h1_prev|h0_new], 128 K/wave ----
    {
      v8f acc = {};
      const _Float16* src = (wave < 8) ? h1f : h0f;
      const int koff = (wave < 8) ? 0 : H_N;
      const int kb = wave * 128;
#pragma unroll
      for (int q = 0; q < 4; ++q) {
        const int k = kb + 32 * q;
        acc = wmma_step(acc, src, k - koff, W1t, 2 * H_N, k, lane, zbuf);
      }
      if (lane < 4) {
        float* pp = partial + wave * 32 + lane * 8;
        *(float4*)(pp)     = make_float4(acc[0], acc[1], acc[2], acc[3]);
        *(float4*)(pp + 4) = make_float4(acc[4], acc[5], acc[6], acc[7]);
      }
    }
    __syncthreads();                                       // B4
    if (tid < 32) {
      const int r = tid >> 2, j = tid & 3;
      float g = biasL[4 + j];
#pragma unroll
      for (int w = 0; w < NWAVES; ++w) g += partial[w * 32 + j * 8 + r];
      Gates[tid] = (j == 1) ? tanhf(g) : 1.f / (1.f + __expf(-g));
    }
    __syncthreads();                                       // B5
    // ---- layer 1 elementwise: c1, h1, stream outs ----
    {
      const size_t ob = (size_t)t * (B_N * H_N) + (size_t)bTile * H_N;
#pragma unroll 4
      for (int i = 0; i < ROWS * H_N / THREADS; ++i) {
        const int e = tid + THREADS * i;
        const int r = e >> 10;
        const float ii = Gates[r * 4 + 0], ch = Gates[r * 4 + 1];
        const float ff = Gates[r * 4 + 2], oo = Gates[r * 4 + 3];
        const float cn = ff * c1[e] + ii * ch;
        c1[e] = cn;
        const float hn = oo * tanhf(cn);
        h1f[e] = (_Float16)hn;
        out[ob + e] = hn;                                  // outs[t, bTile.., :]
        if (t == S_STEPS - 1) {
          out[OUT_HT + (size_t)(B_N + bTile) * H_N + e] = hn;  // hT layer 1
          out[OUT_CT + (size_t)(B_N + bTile) * H_N + e] = cn;  // cT layer 1
        }
      }
    }
    // loop-carried hazards are covered by B0..B3 of the next iteration
  }
}

// ---------------------------------------------------------------------------
extern "C" void kernel_launch(void* const* d_in, const int* in_sizes, int n_in,
                              void* d_out, int out_size, void* d_ws, size_t ws_size,
                              hipStream_t stream)
{
  (void)in_sizes; (void)n_in; (void)out_size; (void)ws_size;

  const float* X  = (const float*)d_in[0];   // [S,B,H]
  const float* Wg = (const float*)d_in[1];   // [L,4,2H]
  const float* bg = (const float*)d_in[2];   // [L,4]
  float* out = (float*)d_out;                // outs | hT | cT (flat f32)
  float* Gx  = (float*)d_ws;                 // [S,B,4] f32 = 2 MB scratch

  gx_kernel<<<dim3((S_STEPS * B_N) / 8), dim3(256), 0, stream>>>(X, Wg, Gx);

  const size_t lds =
      sizeof(float)    * (2 * ROWS * H_N + NWAVES * 32 + 32 + 32 + 8) +
      sizeof(_Float16) * 32 +                                    // zbuf
      sizeof(_Float16) * (2 * ROWS * H_N + 4 * H_N + 8 * H_N);   // 125,280 B

  hipFuncSetAttribute((const void*)lstm_rec_kernel,
                      hipFuncAttributeMaxDynamicSharedMemorySize, (int)lds);
  lstm_rec_kernel<<<dim3(B_N / ROWS), dim3(THREADS), lds, stream>>>(Wg, bg, Gx, out);
}
